// FH_Mamba_15693810499961
// MI455X (gfx1250) — matrix operand order repaired and checked by reading, not compiled, not run on hardware
//
#include <hip/hip_runtime.h>
#include <math.h>

// ---------------------------------------------------------------------------
// FH_Mamba full pipeline for gfx1250 (MI455X, wave32, WMMA + TDM).
// Dense contractions -> v_wmma_f32_16x16x32_f16 (f32 data, f16 operands, f32
// accumulate). Conv weights staged via tensor_load_to_lds (TDM, double
// buffered, s_wait_tensorcnt). All gathers branch-free with 32-bit offsets
// from clamped per-lane base pointers (no per-element 64-bit address math).
// ---------------------------------------------------------------------------

typedef _Float16 half16 __attribute__((ext_vector_type(16)));
typedef float f32x8 __attribute__((ext_vector_type(8)));
typedef unsigned int u32x4 __attribute__((ext_vector_type(4)));
typedef int i32x4 __attribute__((ext_vector_type(4)));
typedef int i32x8 __attribute__((ext_vector_type(8)));

#if defined(__HIP_DEVICE_COMPILE__) && __has_builtin(__builtin_amdgcn_tensor_load_to_lds) && __has_builtin(__builtin_amdgcn_s_wait_tensorcnt)
#define ATHENA_HAS_TDM 1
#else
#define ATHENA_HAS_TDM 0
#endif

__device__ __forceinline__ float dev_silu(float v){ return v/(1.0f+__expf(-v)); }
__device__ __forceinline__ float dev_gelu(float v){ return 0.5f*v*(1.0f+erff(v*0.7071067811865475f)); }
__device__ __forceinline__ float dev_act(float v,int a){ return a==1?dev_silu(v):(a==2?dev_gelu(v):v); }

#if ATHENA_HAS_TDM
// Issue one TDM 2D tile load: 16 rows x tile_cols f32, row stride ldK elems,
// from global `g` into LDS byte offset `lds_off`. OOB rows/cols zero-filled.
__device__ __forceinline__ void tdm_load_tile(const float* g, unsigned lds_off,
                                              int rem_cols, int rem_rows,
                                              int tile_cols, int ldK)
{
    unsigned long long ga = (unsigned long long)(size_t)g;
    u32x4 g0;
    g0[0] = 1u;                                        // count=1 (valid user D#)
    g0[1] = lds_off;                                   // lds_addr (bytes)
    g0[2] = (unsigned)(ga & 0xffffffffu);              // global_addr[31:0]
    g0[3] = (unsigned)((ga >> 32) & 0x01ffffffu)       // global_addr[56:32]
          | (2u << 30);                                // type = 2 ("image")
    i32x8 g1;
    g1[0] = (int)(2u << 16);                           // data_size = 4B
    g1[1] = (int)(((unsigned)rem_cols & 0xffffu) << 16);            // tensor_dim0 lo16
    g1[2] = (int)((((unsigned)rem_cols >> 16) & 0xffffu)
          | (((unsigned)rem_rows & 0xffffu) << 16));                // dim0 hi16 | dim1 lo16
    g1[3] = (int)((((unsigned)rem_rows >> 16) & 0xffffu)
          | (((unsigned)tile_cols & 0xffffu) << 16));               // dim1 hi16 | tile_dim0
    g1[4] = 16;                                        // tile_dim1 = 16 rows (tile_dim2 = 0)
    g1[5] = ldK;                                       // tensor_dim0_stride[31:0] (elements)
    g1[6] = 0;
    g1[7] = 0;
    i32x4 z4 = {0,0,0,0};
#if defined(__clang_major__) && __clang_major__ >= 23
    i32x8 z8 = {0,0,0,0,0,0,0,0};
    __builtin_amdgcn_tensor_load_to_lds(g0, g1, z4, z4, z8, 0);   // clang-23 lane (6-arg)
#else
    __builtin_amdgcn_tensor_load_to_lds(g0, g1, z4, z4, 0);       // ROCm 7.2 clang-22 (5-arg)
#endif
}
#endif

// ---------------- WMMA implicit-GEMM convolution (groups=1) ----------------
// M = Cout (weights OIHW row-major A[M][K]), N = Nimg*Hout*Wo, K = Cin*KH*KW.
// One wave computes a 16(M) x 32(N) output tile; A tiles staged via TDM.
template<int KH, int KW>
__global__ __launch_bounds__(32)
void k_conv_wmma_t(const float* __restrict__ X, const float* __restrict__ Wt,
                   const float* __restrict__ bias, float* __restrict__ Y,
                   int Nimg, int Cin, int Hin, int Win,
                   int Cout, int Hout, int Wo,
                   int stride, int pad, int co_total, int co_off, int act)
{
    constexpr int KHW = KH*KW;
    const int lane = threadIdx.x;
    const int nl = lane & 15;
    const int hi = lane >> 4;
    const int kbA = hi ? 16 : 0;     // f16 A (contiguous k order, same as B): lanes 16-31 start at K+16
    const int kbB = hi ? 16 : 0;     // f16 B (contiguous k order, same as A): lanes 16-31 start at K+16
    const int mtile = blockIdx.y;
    const int Ntot = Nimg*Hout*Wo;
    const int K = Cin*KHW;
    const int KCH = 64;              // TDM chunk: 16 rows x 64 cols f32 = 4KB
    const int imstride = Cin*Hin*Win;    // fits in 32 bits for all shapes here

    int imgoff[2], oh[2], ow[2]; bool nv[2];
#pragma unroll
    for (int s = 0; s < 2; ++s) {
        int n_idx = blockIdx.x*32 + s*16 + nl;
        nv[s] = (n_idx < Ntot);
        int i2 = nv[s] ? n_idx : 0;
        int img = i2/(Hout*Wo);
        int r = i2 - img*(Hout*Wo);
        oh[s] = r/Wo; ow[s] = r - oh[s]*Wo;
        imgoff[s] = img*imstride;
    }
    const int m_a = mtile*16 + nl;
    (void)m_a;

#if ATHENA_HAS_TDM
    extern __shared__ float As[];    // 2 buffers x 16x64 f32 = 8KB dynamic LDS
    const float* wbase = Wt + (size_t)mtile*16*K;
    tdm_load_tile(wbase, 0u, K, Cout - mtile*16, KCH, K);
#else
    // clamped per-lane weight row base (safe even when m_a >= Cout)
    const float* Arow = Wt + (size_t)((m_a < Cout) ? m_a : 0)*K;
    const bool mok = (m_a < Cout);
#endif

    f32x8 acc0 = {0,0,0,0,0,0,0,0};
    f32x8 acc1 = {0,0,0,0,0,0,0,0};

    const int nch = (K + KCH - 1)/KCH;
    for (int c = 0; c < nch; ++c) {
        const int k0 = c*KCH;
#if ATHENA_HAS_TDM
        if (c + 1 < nch) {   // prefetch next weight slab while waiting on current
            tdm_load_tile(wbase + (k0 + KCH), (unsigned)(((c+1)&1)*16*KCH*4),
                          K - (k0 + KCH), Cout - mtile*16, KCH, K);
            __builtin_amdgcn_s_wait_tensorcnt((short)1);
        } else {
            __builtin_amdgcn_s_wait_tensorcnt((short)0);
        }
        asm volatile("" ::: "memory");
        const float* Ab = As + (c & 1)*16*KCH;
#endif
#pragma unroll
        for (int half = 0; half < 2; ++half) {
            const int kk = k0 + half*32;
            if (kk < K) {
                half16 a;
#pragma unroll
                for (int j = 0; j < 16; ++j) {
                    int ka = kk + kbA + j;
#if ATHENA_HAS_TDM
                    // TDM zero-fills OOB rows/cols: unconditional LDS read.
                    a[j] = (_Float16)Ab[nl*KCH + (ka - k0)];
#else
                    bool okA = mok & (ka < K);
                    int kac = (ka < K) ? ka : 0;
                    float raw = Arow[kac];
                    a[j] = (_Float16)(okA ? raw : 0.0f);
#endif
                }
#pragma unroll
                for (int s = 0; s < 2; ++s) {
                    half16 b;
#pragma unroll
                    for (int j = 0; j < 16; ++j) {
                        int kb = kk + kbB + j;
                        bool okk = (kb < K);
                        int kbc = okk ? kb : 0;
                        int ci = kbc / KHW;               // compile-time KHW
                        int rr = kbc - ci*KHW;
                        int kh = rr / KW, kw = rr - kh*KW;
                        int ih = oh[s]*stride - pad + kh;
                        int iw = ow[s]*stride - pad + kw;
                        bool ok = nv[s] & okk
                                & ((unsigned)ih < (unsigned)Hin)
                                & ((unsigned)iw < (unsigned)Win);
                        int idx = imgoff[s] + (ci*Hin + ih)*Win + iw;   // 32-bit
                        idx = ok ? idx : 0;
                        float raw = X[idx];               // unconditional, batched
                        b[j] = (_Float16)(ok ? raw : 0.0f);
                    }
                    if (s == 0)
                        acc0 = __builtin_amdgcn_wmma_f32_16x16x32_f16(false, a, false, b, (short)0, acc0, false, false);
                    else
                        acc1 = __builtin_amdgcn_wmma_f32_16x16x32_f16(false, a, false, b, (short)0, acc1, false, false);
                }
            }
        }
    }
#pragma unroll
    for (int s = 0; s < 2; ++s) {
        if (!nv[s]) continue;
        f32x8 acc = s ? acc1 : acc0;
#pragma unroll
        for (int r = 0; r < 8; ++r) {
            int m = mtile*16 + r + (hi ? 8 : 0);
            if (m < Cout) {
                size_t img = (size_t)(imgoff[s]/imstride);
                size_t o = ((img*co_total + co_off + m)*Hout + oh[s])*Wo + ow[s];
                float v = acc[r] + (bias ? bias[m] : 0.0f);
                Y[o] = dev_act(v, act);
            }
        }
    }
}

// ---------------- WMMA GEMM: C[M,N] = A[M,K] x B[K,N] (+bias +resid) -------
__global__ __launch_bounds__(32)
void k_gemm_wmma(const float* __restrict__ A, const float* __restrict__ Bw,
                 const float* __restrict__ bias, float* __restrict__ C,
                 const float* __restrict__ resid,
                 int M, int N, int K, int act)
{
    const int lane  = threadIdx.x;
    const int mtile = blockIdx.x, ntile = blockIdx.y;
    const int nl = lane & 15, hi = lane >> 4;
    const int m_a = mtile*16 + nl;
    const int n_b = ntile*16 + nl;
    const int kbA = hi ? 16 : 0, kbB = hi ? 16 : 0;
    const bool mok = (m_a < M), nok = (n_b < N);

    // clamped per-lane bases: all subsequent offsets are 32-bit
    const float* Arow = A  + (size_t)(mok ? m_a : 0)*K;
    const float* Bcol = Bw + (nok ? n_b : 0);

    f32x8 acc = {0,0,0,0,0,0,0,0};
    for (int k0 = 0; k0 < K; k0 += 32) {
        half16 a, b;
#pragma unroll
        for (int j = 0; j < 16; ++j) {
            int ka = k0 + kbA + j;
            bool okA = mok & (ka < K);
            int kac = (ka < K) ? ka : 0;
            float ra = Arow[kac];
            a[j] = (_Float16)(okA ? ra : 0.0f);

            int kb = k0 + kbB + j;
            bool okB = nok & (kb < K);
            int kbc = (kb < K) ? kb : 0;
            float rb = Bcol[kbc*N];
            b[j] = (_Float16)(okB ? rb : 0.0f);
        }
        acc = __builtin_amdgcn_wmma_f32_16x16x32_f16(false, a, false, b, (short)0, acc, false, false);
    }
    if (nok) {
#pragma unroll
        for (int r = 0; r < 8; ++r) {
            int m = mtile*16 + r + (hi ? 8 : 0);
            if (m < M) {
                size_t o = (size_t)m*N + n_b;
                float v = acc[r] + (bias ? bias[n_b] : 0.0f);
                if (resid) v += resid[o];
                C[o] = dev_act(v, act);
            }
        }
    }
}

// ---------------- depthwise 3x3 conv (pad 1, stride 1) ---------------------
__global__ void k_dw3(const float* __restrict__ X, const float* __restrict__ Wt,
                      const float* __restrict__ bias, float* __restrict__ Y,
                      int Nimg, int C, int H, int W, int co_total, int co_off)
{
    int gid = blockIdx.x*blockDim.x + threadIdx.x;
    int tot = Nimg*C*H*W;
    if (gid >= tot) return;
    int w = gid % W; int h = (gid/W) % H; int c = (gid/(W*H)) % C; int n = gid/(W*H*C);
    int base = (n*C + c)*H*W;                 // 32-bit
    float acc = bias[c];
#pragma unroll
    for (int kh = 0; kh < 3; ++kh)
#pragma unroll
        for (int kw = 0; kw < 3; ++kw) {
            int ih = h-1+kh, iw = w-1+kw;
            bool ok = ((unsigned)ih < (unsigned)H) & ((unsigned)iw < (unsigned)W);
            int idx = ok ? (base + ih*W + iw) : 0;
            float raw = X[idx];
            acc += Wt[c*9 + kh*3 + kw] * (ok ? raw : 0.0f);
        }
    Y[(((size_t)n*co_total + co_off + c)*H + h)*W + w] = acc;
}

// ---------------- instance norm (per n,c over HW), fused activation --------
__global__ __launch_bounds__(256)
void k_instnorm_act(float* __restrict__ x, int HW, int act)
{
    const int nc = blockIdx.x, tid = threadIdx.x;
    float* p = x + (size_t)nc*HW;
    __shared__ float s1[256], s2[256];
    float a = 0.f, b = 0.f;
    for (int i = tid; i < HW; i += 256) { float v = p[i]; a += v; b += v*v; }
    s1[tid] = a; s2[tid] = b; __syncthreads();
    for (int o = 128; o > 0; o >>= 1) {
        if (tid < o) { s1[tid] += s1[tid+o]; s2[tid] += s2[tid+o]; }
        __syncthreads();
    }
    float mu = s1[0]/HW, var = s2[0]/HW - mu*mu;
    float is = rsqrtf(var + 1e-5f);
    for (int i = tid; i < HW; i += 256) p[i] = dev_act((p[i]-mu)*is, act);
}

// ---------------- layernorm over (H,W) with spatial affine g[hw], b[hw] ----
__global__ __launch_bounds__(256)
void k_ln_hw(const float* __restrict__ x, const float* __restrict__ g,
             const float* __restrict__ bb, float* __restrict__ y, int HW)
{
    const int nc = blockIdx.x, tid = threadIdx.x;
    const float* p = x + (size_t)nc*HW;
    __shared__ float s1[256], s2[256];
    float a = 0.f, b = 0.f;
    for (int i = tid; i < HW; i += 256) { float v = p[i]; a += v; b += v*v; }
    s1[tid] = a; s2[tid] = b; __syncthreads();
    for (int o = 128; o > 0; o >>= 1) {
        if (tid < o) { s1[tid] += s1[tid+o]; s2[tid] += s2[tid+o]; }
        __syncthreads();
    }
    float mu = s1[0]/HW, var = s2[0]/HW - mu*mu;
    float is = rsqrtf(var + 1e-5f);
    for (int i = tid; i < HW; i += 256)
        y[(size_t)nc*HW + i] = (p[i]-mu)*is*g[i] + bb[i];
}

// ---------------- layernorm over last dim (C=16) ---------------------------
__global__ void k_ln16(const float* __restrict__ x, const float* __restrict__ g,
                       const float* __restrict__ b, float* __restrict__ y, int rows)
{
    int r = blockIdx.x*blockDim.x + threadIdx.x;
    if (r >= rows) return;
    const float* p = x + (size_t)r*16;
    float mu = 0.f;
#pragma unroll
    for (int c = 0; c < 16; ++c) mu += p[c];
    mu *= (1.0f/16.0f);
    float var = 0.f;
#pragma unroll
    for (int c = 0; c < 16; ++c) { float d = p[c]-mu; var += d*d; }
    var *= (1.0f/16.0f);
    float is = rsqrtf(var + 1e-5f);
#pragma unroll
    for (int c = 0; c < 16; ++c) y[(size_t)r*16 + c] = (p[c]-mu)*is*g[c] + b[c];
}

// ---------------- per-channel affine + gelu (bn path) ----------------------
__global__ void k_chan_affine_gelu(float* __restrict__ x, const float* __restrict__ g,
                                   const float* __restrict__ b, int C, int HW, int total)
{
    int gid = blockIdx.x*blockDim.x + threadIdx.x;
    if (gid >= total) return;
    int c = (gid / HW) % C;
    float v = x[gid] * (g[c] * 0.9999950000374997f) + b[c];   // g/sqrt(1+eps)
    x[gid] = dev_gelu(v);
}

// ---------------- Haar wavelet forward: (B,C,H,W)->(B,4C,H/2,W/2) ----------
__global__ void k_wt_fwd(const float* __restrict__ x, float* __restrict__ y,
                         int Bb, int C, int H, int W)
{
    int h2 = H>>1, w2 = W>>1;
    int tot = Bb*C*h2*w2;
    int gid = blockIdx.x*blockDim.x + threadIdx.x;
    if (gid >= tot) return;
    int j = gid % w2; int i = (gid/w2) % h2; int c = (gid/(w2*h2)) % C; int b = gid/(w2*h2*C);
    size_t base = (((size_t)b*C + c)*H + 2*i)*W + 2*j;
    float x00 = x[base], x01 = x[base+1], x10 = x[base+W], x11 = x[base+W+1];
    size_t ck = (size_t)C*h2*w2;
    size_t ob = ((size_t)b*4*C)*h2*w2 + (size_t)c*h2*w2 + (size_t)i*w2 + j;
    y[ob       ] = 0.5f*(x00+x01+x10+x11);   // LL
    y[ob +   ck] = 0.5f*(x00+x01-x10-x11);   // LH
    y[ob + 2*ck] = 0.5f*(x00-x01+x10-x11);   // HL
    y[ob + 3*ck] = 0.5f*(x00-x01-x10+x11);   // HH
}

// ---------------- Haar wavelet inverse: (B,4C,h,w)->(B,C,2h,2w) ------------
__global__ void k_wt_inv(const float* __restrict__ y, float* __restrict__ out,
                         int Bb, int C, int h, int w)
{
    int tot = Bb*C*h*w;
    int gid = blockIdx.x*blockDim.x + threadIdx.x;
    if (gid >= tot) return;
    int j = gid % w; int i = (gid/w) % h; int c = (gid/(w*h)) % C; int b = gid/(w*h*C);
    size_t ck = (size_t)C*h*w;
    size_t ib = ((size_t)b*4*C)*h*w + (size_t)c*h*w + (size_t)i*w + j;
    float v0 = y[ib], v1 = y[ib+ck], v2 = y[ib+2*ck], v3 = y[ib+3*ck];
    size_t ob = (((size_t)b*C + c)*(2*h) + 2*i)*(2*w) + 2*j;
    int W2 = 2*w;
    out[ob     ] = 0.5f*(v0+v1+v2+v3);
    out[ob+1   ] = 0.5f*(v0+v1-v2-v3);
    out[ob+W2  ] = 0.5f*(v0-v1+v2-v3);
    out[ob+W2+1] = 0.5f*(v0-v1-v2+v3);
}

// ---------------- pixel shuffle: (N,4C,H,W) -> (N,C,2H,2W) -----------------
__global__ void k_ps(const float* __restrict__ x, float* __restrict__ y,
                     int Nimg, int C, int H, int W)
{
    int tot = Nimg*C*4*H*W;
    int gid = blockIdx.x*blockDim.x + threadIdx.x;
    if (gid >= tot) return;
    int W2 = 2*W, H2 = 2*H;
    int ow = gid % W2; int oh = (gid/W2) % H2; int c = (gid/(W2*H2)) % C; int n = gid/(W2*H2*C);
    int p = oh & 1, q = ow & 1, i = oh >> 1, j = ow >> 1;
    y[gid] = x[(((size_t)n*4*C + c*4 + p*2 + q)*H + i)*W + j];
}

// ---------------- elementwise add ------------------------------------------
__global__ void k_add(const float* __restrict__ a, const float* __restrict__ b,
                      float* __restrict__ y, int n)
{
    int gid = blockIdx.x*blockDim.x + threadIdx.x;
    if (gid < n) y[gid] = a[gid] + b[gid];
}

// ---------------- Hilbert-like permutation inverse (rank of Morton key) ----
__device__ __forceinline__ long long morton_code(int t, int h, int w)
{
    long long m = 0;
#pragma unroll
    for (int b = 0; b < 6; ++b) {
        m |= (long long)((h>>b)&1) << (3*b);
        m |= (long long)((w>>b)&1) << (3*b+1);
        m |= (long long)((t>>b)&1) << (3*b+2);
    }
    return m;
}
__global__ void k_perm_inv(int* __restrict__ inv, int T_, int H_, int W_)
{
    int L = T_*H_*W_;
    int i = blockIdx.x*blockDim.x + threadIdx.x;
    if (i >= L) return;
    int HW = H_*W_;
    int t = i/HW; int r = i - t*HW; int h = r/W_; int w = r - h*W_;
    long long mi = morton_code(t, h, w);
    int rank = 0;
    for (int j = 0; j < L; ++j) {
        int tj = j/HW; int rj = j - tj*HW; int hj = rj/W_; int wj = rj - hj*W_;
        long long mj = morton_code(tj, hj, wj);
        rank += (mj < mi) ? 1 : 0;
    }
    inv[i] = rank;
}

// tok[b, inv[i], c] = x[b,t,c,h,w] + wout[b,t,c,h,w]
__global__ void k_make_tokens(const float* __restrict__ x, const float* __restrict__ wo,
                              const int* __restrict__ inv, float* __restrict__ tok,
                              int Bb, int L, int C, int T_, int HH, int WW)
{
    int tot = Bb*L*C;
    int gid = blockIdx.x*blockDim.x + threadIdx.x;
    if (gid >= tot) return;
    int c = gid % C; int i = (gid/C) % L; int b = gid/(C*L);
    int HW = HH*WW;
    int t = i/HW; int r = i - t*HW; int h = r/WW; int w = r - h*WW;
    size_t src = ((((size_t)b*T_ + t)*C + c)*HH + h)*WW + w;
    tok[((size_t)b*L + inv[i])*C + c] = x[src] + wo[src];
}

// m[b,t,c,h,w] = ot[b, inv[i], c]
__global__ void k_untoken(const float* __restrict__ ot, const int* __restrict__ inv,
                          float* __restrict__ m, int Bb, int L, int C, int T_, int HH, int WW)
{
    int tot = Bb*L*C;
    int gid = blockIdx.x*blockDim.x + threadIdx.x;
    if (gid >= tot) return;
    int c = gid % C; int i = (gid/C) % L; int b = gid/(C*L);
    int HW = HH*WW;
    int t = i/HW; int r = i - t*HW; int h = r/WW; int w = r - h*WW;
    size_t dst = ((((size_t)b*T_ + t)*C + c)*HH + h)*WW + w;
    m[dst] = ot[((size_t)b*L + inv[i])*C + c];
}

// causal depthwise conv1d over sequence (K=4) reading x-half of xz, + silu
__global__ void k_conv1d_silu(const float* __restrict__ xz, const float* __restrict__ Wt,
                              const float* __restrict__ bias, float* __restrict__ xs,
                              int Bb, int L, int D)
{
    int tot = Bb*L*D;
    int gid = blockIdx.x*blockDim.x + threadIdx.x;
    if (gid >= tot) return;
    int d = gid % D; int l = (gid/D) % L; int b = gid/(D*L);
    int base = (b*L)*32 + d;                 // 32-bit
    float acc = bias[d];
#pragma unroll
    for (int k = 0; k < 4; ++k) {
        int t = l - 3 + k;
        bool ok = (t >= 0);
        int idx = ok ? (base + t*32) : 0;
        float raw = xz[idx];
        acc += Wt[d*4 + k] * (ok ? raw : 0.0f);
    }
    xs[gid] = dev_silu(acc);
}

// split xdbl(.,33) -> dt=softplus(dtr*Wdt+bdt), Bm, Cm
__global__ void k_prep_scan(const float* __restrict__ xdbl, const float* __restrict__ Wdt,
                            const float* __restrict__ bdt, float* __restrict__ dt,
                            float* __restrict__ Bm, float* __restrict__ Cm, int tokens)
{
    int gid = blockIdx.x*blockDim.x + threadIdx.x;
    if (gid >= tokens*16) return;
    int tok = gid >> 4, j = gid & 15;
    const float* row = xdbl + (size_t)tok*33;
    float v = row[0]*Wdt[j] + bdt[j];
    dt[gid] = (v > 20.f) ? v : log1pf(__expf(v));
    Bm[gid] = row[1 + j];
    Cm[gid] = row[17 + j];
}

// selective scan: block per batch, thread (d,n), LDS-chunked, 16-lane reduce
__global__ __launch_bounds__(256)
void k_scan(const float* __restrict__ xs, const float* __restrict__ dt,
            const float* __restrict__ Bm, const float* __restrict__ Cm,
            const float* __restrict__ A_log, const float* __restrict__ Dp,
            float* __restrict__ y, int L)
{
    const int b = blockIdx.x, tid = threadIdx.x;
    const int d = tid >> 4, n = tid & 15;
    const float Av = -__expf(A_log[d*16 + n]);
    const float Dv = Dp[d];
    float h = 0.0f;
    __shared__ float s_xs[64*16], s_dt[64*16], s_B[64*16], s_C[64*16];
    for (int t0 = 0; t0 < L; t0 += 64) {
        __syncthreads();
        const size_t base = ((size_t)b*L + t0)*16;
        for (int i = tid; i < 1024; i += 256) {
            s_xs[i] = xs[base+i]; s_dt[i] = dt[base+i];
            s_B[i]  = Bm[base+i]; s_C[i]  = Cm[base+i];
        }
        __syncthreads();
        for (int tt = 0; tt < 64; ++tt) {
            float dtv = s_dt[tt*16 + d];
            float xv  = s_xs[tt*16 + d];
            float dA  = __expf(dtv*Av);
            h = dA*h + dtv*s_B[tt*16 + n]*xv;
            float p = h*s_C[tt*16 + n];
            p += __shfl_xor(p, 1, 16);
            p += __shfl_xor(p, 2, 16);
            p += __shfl_xor(p, 4, 16);
            p += __shfl_xor(p, 8, 16);
            if (n == 0) y[((size_t)b*L + t0 + tt)*16 + d] = p + xv*Dv;
        }
    }
}

// y *= silu(z) where z = xz[..,16:32]
__global__ void k_mul_silu_z(float* __restrict__ y, const float* __restrict__ xz, int tokens)
{
    int gid = blockIdx.x*blockDim.x + threadIdx.x;
    if (gid >= tokens*16) return;
    int tok = gid >> 4, d = gid & 15;
    float z = xz[(size_t)tok*32 + 16 + d];
    y[gid] *= dev_silu(z);
}

// ===========================================================================
extern "C" void kernel_launch(void* const* d_in, const int* in_sizes, int n_in,
                              void* d_out, int out_size, void* d_ws, size_t ws_size,
                              hipStream_t stream)
{
    (void)in_sizes; (void)n_in; (void)out_size; (void)ws_size;
    auto P = [&](int i) { return (const float*)d_in[i]; };

    // dims
    const int Bb = 4, T_ = 8, CIN = 3, H0 = 128, W0 = 128;
    const int BT = Bb*T_;                 // 32 images
    const int L = T_*32*32;               // 8192 tokens per batch
    const int TOKS = Bb*L;                // 32768 tokens total

    // parameter indices (setup_inputs dict order, leaves depth-first)
    const int ENC0 = 2;                   // 4 dbc blocks x 8 tensors
    const int RIW = 34, RIB = 35;
    const int HID0 = 36;                  // 3 meta blocks x 31 tensors
    const int ROW = 129, ROB = 130;
    const int DEC = 131;

    // deterministic bump allocator over d_ws (floats)
    float* ws = (float*)d_ws;
    size_t cur = 0;
    auto alloc = [&](size_t n) { float* p = ws + cur; cur += (n + 63) & ~(size_t)63; return p; };

    float* CAT  = alloc(33554432);  // concat / decoder ping (up to 32x64x128x128)
    float* ENC1 = alloc(16777216);  // skip connection (32,32,128,128)
    float* S1   = alloc(16777216);
    float* S2   = alloc(16777216);
    float* Z    = alloc(524288);    // running hidden state (4,8,16,32,32)
    float* WB1  = alloc(524288);    // (4,512,16,16)
    float* WB2  = alloc(524288);
    float* WOUT = alloc(524288);
    float* MB   = alloc(524288);    // mamba output grid
    float* FM   = alloc(524288);
    float* FC1  = alloc(524288);
    float* FC2  = alloc(524288);
    float* TOK  = alloc(524288);
    float* LN   = alloc(524288);
    float* XZ   = alloc(1048576);   // (32768,32)
    float* XS   = alloc(524288);
    float* XDBL = alloc(1081344);   // (32768,33)
    float* DT   = alloc(524288);
    float* BMb  = alloc(524288);
    float* CMb  = alloc(524288);
    float* YB   = alloc(524288);
    float* OT   = alloc(524288);
    float* H2   = alloc(524288);
    float* MLP1B= alloc(622592);    // (32768,19)
    float* OT2  = alloc(524288);
    float* HIDB = alloc(1048576);   // (32,32,32,32)
    int*   INV  = (int*)alloc(8192);

    auto conv = [&](const float* X, const float* Wt, const float* bias, float* Y,
                    int Nimg,int Cin,int Hin,int Win,int Cout,int KH,int KW,int s,int pad,
                    int co_total,int co_off,int act) {
        int Hout = (Hin + 2*pad - KH)/s + 1;
        int Wout = (Win + 2*pad - KW)/s + 1;
        int Ntot = Nimg*Hout*Wout;
        dim3 g((unsigned)((Ntot+31)/32), (unsigned)((Cout+15)/16));
        const size_t smem = 2*16*64*sizeof(float);   // TDM double buffer
        if (KH == 1)
            k_conv_wmma_t<1,1><<<g, 32, smem, stream>>>(X, Wt, bias, Y, Nimg, Cin, Hin, Win,
                                                        Cout, Hout, Wout, s, pad, co_total, co_off, act);
        else if (KH == 3)
            k_conv_wmma_t<3,3><<<g, 32, smem, stream>>>(X, Wt, bias, Y, Nimg, Cin, Hin, Win,
                                                        Cout, Hout, Wout, s, pad, co_total, co_off, act);
        else
            k_conv_wmma_t<15,15><<<g, 32, smem, stream>>>(X, Wt, bias, Y, Nimg, Cin, Hin, Win,
                                                          Cout, Hout, Wout, s, pad, co_total, co_off, act);
    };
    auto gemm = [&](const float* A, const float* Bw, const float* bias, float* C,
                    const float* resid, int M, int N, int K, int act) {
        dim3 g((unsigned)((M+15)/16), (unsigned)((N+15)/16));
        k_gemm_wmma<<<g, 32, 0, stream>>>(A, Bw, bias, C, resid, M, N, K, act);
    };
    auto elems = [&](int n) { return dim3((unsigned)((n+255)/256)); };

    // -------------------- encoder (4 DBC blocks) --------------------
    auto dbc = [&](int pb, const float* xin, float* out, float* tmp,
                   int Cin, int Hin, int Win, int s) {
        conv(xin, P(pb+0), P(pb+1), CAT, BT, Cin, Hin, Win, 32, 1,1,1,0, 64, 0, 0);   // b1
        conv(xin, P(pb+2), P(pb+3), tmp, BT, Cin, Hin, Win, 32, 1,1,1,0, 32, 0, 0);   // b2c
        int tot = BT*32*Hin*Win;
        k_dw3<<<elems(tot), 256, 0, stream>>>(tmp, P(pb+4), P(pb+5), CAT, BT, 32, Hin, Win, 64, 32); // b2d
        conv(CAT, P(pb+6), P(pb+7), out, BT, 64, Hin, Win, 32, 3,3,s,1, 32, 0, 0);    // fb
        int Ho = (Hin+2-3)/s + 1, Wo = (Win+2-3)/s + 1;
        k_instnorm_act<<<BT*32, 256, 0, stream>>>(out, Ho*Wo, 1);                     // silu(instnorm)
    };
    dbc(ENC0+0,  (const float*)d_in[0], ENC1, S1, CIN, H0, W0, 1);  // -> (32,32,128,128)
    dbc(ENC0+8,  ENC1, S2, S1, 32, 128, 128, 2);                    // -> (32,32,64,64)
    dbc(ENC0+16, S2,   S1, S1, 32,  64,  64, 1);                    // tmp freed before fb writes S1
    dbc(ENC0+24, S1,   S2, S2, 32,  64,  64, 2);                    // -> (32,32,32,32)

    // ri projection: 1x1 "einsum"  (32,32,32,32) -> Z (4,8,16,32,32)
    conv(S2, P(RIW), P(RIB), Z, BT, 32, 32, 32, 16, 1,1,1,0, 16, 0, 0);

    // permutation (same for all blocks)
    k_perm_inv<<<elems(L), 256, 0, stream>>>(INV, T_, 32, 32);

    // -------------------- 3 meta blocks --------------------
    for (int blk = 0; blk < 3; ++blk) {
        const int pb = HID0 + blk*31;   // mamba params at pb+0..18

        // wavelet branch
        k_wt_fwd<<<elems(Bb*128*16*16), 256, 0, stream>>>(Z, WB1, Bb, 128, 32, 32);
        k_dw3<<<elems(Bb*512*16*16), 256, 0, stream>>>(WB1, P(pb+19), P(pb+20), WB2, Bb, 512, 16, 16, 512, 0);
        k_instnorm_act<<<Bb*512, 256, 0, stream>>>(WB2, 256, 2);
        conv(WB2, P(pb+21), P(pb+22), WB1, Bb, 512, 16, 16, 512, 1,1,1,0, 512, 0, 0);
        k_chan_affine_gelu<<<elems(Bb*512*256), 256, 0, stream>>>(WB1, P(pb+23), P(pb+24), 512, 256, Bb*512*256);
        k_wt_inv<<<elems(Bb*128*16*16), 256, 0, stream>>>(WB1, WOUT, Bb, 128, 16, 16);

        // mamba branch
        k_make_tokens<<<elems(TOKS*16), 256, 0, stream>>>(Z, WOUT, INV, TOK, Bb, L, 16, T_, 32, 32);
        k_ln16<<<elems(TOKS), 256, 0, stream>>>(TOK, P(pb+0), P(pb+1), LN, TOKS);
        gemm(LN, P(pb+2), P(pb+3), XZ, nullptr, TOKS, 32, 16, 0);                  // W_in
        k_conv1d_silu<<<elems(TOKS*16), 256, 0, stream>>>(XZ, P(pb+4), P(pb+5), XS, Bb, L, 16);
        gemm(XS, P(pb+6), nullptr, XDBL, nullptr, TOKS, 33, 16, 0);                // W_x
        k_prep_scan<<<elems(TOKS*16), 256, 0, stream>>>(XDBL, P(pb+7), P(pb+8), DT, BMb, CMb, TOKS);
        k_scan<<<Bb, 256, 0, stream>>>(XS, DT, BMb, CMb, P(pb+9), P(pb+10), YB, L);
        k_mul_silu_z<<<elems(TOKS*16), 256, 0, stream>>>(YB, XZ, TOKS);
        gemm(YB, P(pb+11), P(pb+12), OT, TOK, TOKS, 16, 16, 0);                    // W_out + res
        k_ln16<<<elems(TOKS), 256, 0, stream>>>(OT, P(pb+13), P(pb+14), H2, TOKS);
        gemm(H2, P(pb+15), P(pb+16), MLP1B, nullptr, TOKS, 19, 16, 2);             // gelu(W_m1)
        gemm(MLP1B, P(pb+17), P(pb+18), OT2, OT, TOKS, 16, 19, 0);                 // W_m2 + res
        k_untoken<<<elems(TOKS*16), 256, 0, stream>>>(OT2, INV, MB, Bb, L, 16, T_, 32, 32);

        // fusion branch: layernorm_hw -> 15x15 convs (WMMA+TDM) -> residual add
        k_ln_hw<<<Bb*128, 256, 0, stream>>>(MB, P(pb+25), P(pb+26), FM, 1024);
        conv(FM,  P(pb+27), P(pb+28), FC1, Bb, 128, 32, 32, 128, 15,15,1,7, 128, 0, 0);
        k_instnorm_act<<<Bb*128, 256, 0, stream>>>(FC1, 1024, 2);
        conv(FC1, P(pb+29), P(pb+30), FC2, Bb, 128, 32, 32, 128, 15,15,1,7, 128, 0, 0);
        k_instnorm_act<<<Bb*128, 256, 0, stream>>>(FC2, 1024, 2);
        k_add<<<elems(524288), 256, 0, stream>>>(FC2, MB, Z, 524288);
    }

    // ro projection: (4,8,16,32,32) -> hid (32,32,32,32)
    conv(Z, P(ROW), P(ROB), HIDB, BT, 16, 32, 32, 32, 1,1,1,0, 32, 0, 0);

    // -------------------- decoder --------------------
    conv(HIDB, P(DEC+0), P(DEC+1), CAT, BT, 32, 32, 32, 128, 3,3,1,1, 128, 0, 0);  // up0
    k_ps<<<elems(BT*32*64*64), 256, 0, stream>>>(CAT, S1, BT, 32, 32, 32);         // -> (32,32,64,64)
    conv(S1, P(DEC+2), P(DEC+3), CAT, BT, 32, 64, 64, 32, 3,3,1,1, 32, 0, 0);      // c1
    conv(CAT, P(DEC+4), P(DEC+5), S2, BT, 32, 64, 64, 128, 3,3,1,1, 128, 0, 0);    // up2
    k_ps<<<elems(BT*32*128*128), 256, 0, stream>>>(S2, CAT, BT, 32, 64, 64);       // -> (32,32,128,128)
    k_add<<<elems(16777216), 256, 0, stream>>>(CAT, ENC1, CAT, 16777216);          // + enc1
    conv(CAT, P(DEC+6), P(DEC+7), S2, BT, 32, 128, 128, 32, 3,3,1,1, 32, 0, 0);    // c3
    conv(S2, P(DEC+8), P(DEC+9), (float*)d_out, BT, 32, 128, 128, 3, 1,1,1,0, 3, 0, 0); // ro -> output
}
